// GraphConvolution_83571473645637
// MI455X (gfx1250) — compile-verified, matrix operand order checked by reading, not run on hardware
//
#include <hip/hip_runtime.h>
#include <hip/hip_bf16.h>
#include <math.h>

// Problem constants (from reference): B=4, N=512, D=512, relations 1..40.
#define BB 4
#define NN 512
#define DD 512
#define BND (BB * NN * DD) /* 1,048,576 floats = 4 MB */
#define RCHUNK 8           /* relations per msg chunk -> 32 MB scratch */

typedef __attribute__((ext_vector_type(16))) __bf16 v16bf;
typedef __attribute__((ext_vector_type(8)))  float  v8f;

// Load one bf16 A/B fragment (16 values per lane) for the 16x16x32 WMMA.
// Layout (ISA 7.12.2, 16-bit A 16x32; B mirrors with lane&15 = column):
//   elements 0..7  = K[kk + 8*half + 0..7]
//   elements 8..15 = K[kk + 16 + 8*half + 0..7]
__device__ __forceinline__ v16bf load_frag_bf16(const float* __restrict__ row,
                                                int kk, int half) {
  const float4* p0 = (const float4*)(row + kk + 8 * half);
  const float4* p1 = (const float4*)(row + kk + 16 + 8 * half);
  float4 f0 = p0[0], f1 = p0[1], f2 = p1[0], f3 = p1[1];
  v16bf v;
  v[0]  = (__bf16)f0.x; v[1]  = (__bf16)f0.y; v[2]  = (__bf16)f0.z; v[3]  = (__bf16)f0.w;
  v[4]  = (__bf16)f1.x; v[5]  = (__bf16)f1.y; v[6]  = (__bf16)f1.z; v[7]  = (__bf16)f1.w;
  v[8]  = (__bf16)f2.x; v[9]  = (__bf16)f2.y; v[10] = (__bf16)f2.z; v[11] = (__bf16)f2.w;
  v[12] = (__bf16)f3.x; v[13] = (__bf16)f3.y; v[14] = (__bf16)f3.z; v[15] = (__bf16)f3.w;
  return v;
}

// ---------------------------------------------------------------------------
// GEMM: Y[M,Nout] = X[M,K] * Wt[Nout,K]^T  (torch-Linear layout: Wt is [out,in])
// Register-blocked 32x32 output per wave32: 2 A fragments x 2 B fragments per
// K-step -> 4 independent v_wmma_f32_16x16x32_bf16 accumulator chains.
// C/D layout: VGPR r holds row (8*(lane>>4) + r), col = lane&15.
// ---------------------------------------------------------------------------
__global__ __launch_bounds__(256) void gemm_bf16_wmma(
    const float* __restrict__ X, const float* __restrict__ Wt,
    float* __restrict__ Y, int M, int K, int Nout) {
  const int lane = threadIdx.x & 31;
  const int wave = threadIdx.x >> 5;
  const int tiles_n2 = Nout >> 5;               // 32-wide macro tiles
  const int t = blockIdx.x * 8 + wave;
  const int tile_m = t / tiles_n2;
  const int tile_n = t % tiles_n2;
  if (tile_m * 32 >= M) return;

  const int rsel = lane & 15;   // A: row-in-16, B: col-in-16, C: col
  const int half = lane >> 4;

  const float* xrow0 = X  + (size_t)(tile_m * 32 + rsel) * K;
  const float* xrow1 = xrow0 + (size_t)16 * K;
  const float* wrow0 = Wt + (size_t)(tile_n * 32 + rsel) * K;
  const float* wrow1 = wrow0 + (size_t)16 * K;

  v8f c00 = {}, c01 = {}, c10 = {}, c11 = {};
  for (int kk = 0; kk < K; kk += 32) {
    v16bf a0 = load_frag_bf16(xrow0, kk, half);
    v16bf a1 = load_frag_bf16(xrow1, kk, half);
    v16bf b0 = load_frag_bf16(wrow0, kk, half);
    v16bf b1 = load_frag_bf16(wrow1, kk, half);
    c00 = __builtin_amdgcn_wmma_f32_16x16x32_bf16(false, a0, false, b0,
                                                  (short)0, c00, false, false);
    c01 = __builtin_amdgcn_wmma_f32_16x16x32_bf16(false, a0, false, b1,
                                                  (short)0, c01, false, false);
    c10 = __builtin_amdgcn_wmma_f32_16x16x32_bf16(false, a1, false, b0,
                                                  (short)0, c10, false, false);
    c11 = __builtin_amdgcn_wmma_f32_16x16x32_bf16(false, a1, false, b1,
                                                  (short)0, c11, false, false);
  }

  const int col0 = tile_n * 32 + rsel;
#pragma unroll
  for (int r = 0; r < 8; ++r) {
    int m0 = tile_m * 32 + 8 * half + r;
    int m1 = m0 + 16;
    Y[(size_t)m0 * Nout + col0]      = c00[r];
    Y[(size_t)m0 * Nout + col0 + 16] = c01[r];
    Y[(size_t)m1 * Nout + col0]      = c10[r];
    Y[(size_t)m1 * Nout + col0 + 16] = c11[r];
  }
}

// ---------------------------------------------------------------------------
// msg[rl][b,j,d] = (H + bias[rl]) * sigmoid(G + gbias[rl]) for a relation chunk.
// ---------------------------------------------------------------------------
__global__ __launch_bounds__(256) void msg_kernel(
    const float* __restrict__ H, const float* __restrict__ G,
    const float* __restrict__ bias, const float* __restrict__ gbias,
    float* __restrict__ MSG, int nrel) {
  long tid = (long)blockIdx.x * 256 + threadIdx.x;
  long total = (long)nrel * BND;
  if (tid >= total) return;
  int rl = (int)(tid / BND);
  long rem = tid - (long)rl * BND;
  int d = (int)(rem & (DD - 1));
  float h = H[rem] + bias[rl * DD + d];
  float g = G[rem] + gbias[rl * DD + d];
  MSG[tid] = h * (1.0f / (1.0f + __expf(-g)));
}

// ---------------------------------------------------------------------------
// Gather-accumulate: for block (b,i), out[b,i,:] += sum_j msg[adj-sel(b,i,j)].
// Relation row staged in LDS; per-j test scalarized via readfirstlane so the
// whole wave skips non-matching rows on the SALU branch. Lookahead prefetch of
// the next matching msg row (global_prefetch_b8).
// ---------------------------------------------------------------------------
__global__ __launch_bounds__(256) void gather_kernel(
    const float* __restrict__ MSG, const int* __restrict__ adj,
    float* __restrict__ out, int lo, int hi, int transposed) {
  __shared__ int rel[NN];
  const int b = blockIdx.x >> 9;        // / NN
  const int i = blockIdx.x & (NN - 1);  // % NN
  const int tid = threadIdx.x;

  for (int j = tid; j < NN; j += 256) {
    rel[j] = transposed ? adj[((size_t)b * NN + j) * NN + i]
                        : adj[((size_t)b * NN + i) * NN + j];
  }
  __syncthreads();

  const int d0 = tid * 2;  // each thread owns two consecutive d's
  float2 acc = make_float2(0.0f, 0.0f);

  for (int j = 0; j < NN; ++j) {
    int v = __builtin_amdgcn_readfirstlane(rel[j]);
    if (v >= lo && v < hi) {
      // prefetch the next matching row a few iterations ahead
      int jp = j + 6;
      if (jp < NN) {
        int vp = __builtin_amdgcn_readfirstlane(rel[jp]);
        if (vp >= lo && vp < hi) {
          const float* pf = MSG + (size_t)(vp - lo) * BND +
                            ((size_t)b * NN + jp) * DD + d0;
          __builtin_prefetch(pf, 0, 0);
        }
      }
      const float2* row = (const float2*)(MSG + (size_t)(v - lo) * BND +
                                          ((size_t)b * NN + j) * DD);
      float2 m = row[tid];
      acc.x += m.x;
      acc.y += m.y;
    }
  }

  float2* o = (float2*)(out + ((size_t)b * NN + i) * DD);
  float2 cur = o[tid];
  cur.x += acc.x;
  cur.y += acc.y;
  o[tid] = cur;
}

__global__ __launch_bounds__(256) void init_kernel(float* __restrict__ out) {
  out[(size_t)blockIdx.x * 256 + threadIdx.x] = 0.0f;
}

__global__ __launch_bounds__(256) void relu_kernel(float* __restrict__ out) {
  size_t i = (size_t)blockIdx.x * 256 + threadIdx.x;
  out[i] = fmaxf(out[i], 0.0f);
}

// ---------------------------------------------------------------------------
extern "C" void kernel_launch(void* const* d_in, const int* in_sizes, int n_in,
                              void* d_out, int out_size, void* d_ws, size_t ws_size,
                              hipStream_t stream) {
  (void)in_sizes; (void)n_in; (void)out_size; (void)ws_size;
  const float* node   = (const float*)d_in[0];
  // d_in[1] = node_mask (all ones, unused by reference math)
  const int*   adj    = (const int*)d_in[2];
  const float* W_dir  = (const float*)d_in[3];
  const float* W_gate = (const float*)d_in[4];
  const float* bias1  = (const float*)d_in[5];
  const float* bias2  = (const float*)d_in[6];
  const float* bias3  = (const float*)d_in[7];
  const float* gbias1 = (const float*)d_in[8];
  const float* gbias2 = (const float*)d_in[9];
  const float* gbias3 = (const float*)d_in[10];
  float* out = (float*)d_out;

  float* H   = (float*)d_ws;     // 4 MB
  float* G   = H + BND;          // 4 MB
  float* MSG = G + BND;          // up to 32 MB (RCHUNK relations)

  init_kernel<<<BND / 256, 256, 0, stream>>>(out);

  struct DirCfg {
    int w; const float* bias; const float* gbias; int rlo, rhi, transposed;
  };
  DirCfg dirs[3] = {
      {1, bias2, gbias2, 2, 41, 0},  // forward adj, relations 2..40
      {2, bias3, gbias3, 2, 41, 1},  // transposed adj, relations 2..40
      {0, bias1, gbias1, 1, 2, 1},   // transposed adj, relation {1}
  };

  // 32x32 macro-tiles: (2048/32) * (512/32) = 64*16 = 1024 wave-tiles, 8/block.
  const int gemm_blocks = ((BB * NN / 32) * (DD / 32)) / 8;  // 128 blocks

  for (int di = 0; di < 3; ++di) {
    const float* Wd = W_dir  + (size_t)dirs[di].w * DD * DD;
    const float* Wg = W_gate + (size_t)dirs[di].w * DD * DD;
    gemm_bf16_wmma<<<gemm_blocks, 256, 0, stream>>>(node, Wd, H, BB * NN, DD, DD);
    gemm_bf16_wmma<<<gemm_blocks, 256, 0, stream>>>(H, Wg, G, BB * NN, DD, DD);

    const int rbase = dirs[di].rlo;
    for (int lo = dirs[di].rlo; lo < dirs[di].rhi; lo += RCHUNK) {
      int hi = (lo + RCHUNK < dirs[di].rhi) ? (lo + RCHUNK) : dirs[di].rhi;
      int nrel = hi - lo;
      const float* bptr  = dirs[di].bias  + (size_t)(lo - rbase) * DD;
      const float* gbptr = dirs[di].gbias + (size_t)(lo - rbase) * DD;
      long total = (long)nrel * BND;
      unsigned mblocks = (unsigned)((total + 255) / 256);
      msg_kernel<<<mblocks, 256, 0, stream>>>(H, G, bptr, gbptr, MSG, nrel);
      gather_kernel<<<BB * NN, 256, 0, stream>>>(MSG, adj, out, lo, hi,
                                                 dirs[di].transposed);
    }
  }

  relu_kernel<<<BND / 256, 256, 0, stream>>>(out);
}
